// HWC_SpatialAttention_73650099191961
// MI455X (gfx1250) — compile-verified, matrix operand order checked by reading, not run on hardware
//
#include <hip/hip_runtime.h>

// Problem dims (fixed by reference): B=4,S=8 -> 32 slices; H*W=1024; C1=C2=C=256.
#define NSLICE 32
#define HW     1024
#define CCH    256
#define SCALE  0.0625f   // 1/sqrt(256)

typedef __attribute__((ext_vector_type(16))) __bf16 v16bf;
typedef __attribute__((ext_vector_type(8)))  float  v8f;

union FragBF { v16bf v; uint4 u[2]; };

// Load one WMMA operand fragment: lane holds 16 contiguous bf16 (32 bytes).
// Works for both A fragments (lane = row M, K contiguous) and B fragments
// (lane = col N, K contiguous) because B-frag == B^T in A layout.
__device__ __forceinline__ v16bf load_frag(const __bf16* p) {
  FragBF f;
  f.u[0] = *(const uint4*)(p);
  f.u[1] = *(const uint4*)(p + 8);
  return f.v;
}

__device__ __forceinline__ v8f wmma_bf16(v16bf a, v16bf b, v8f c) {
  return __builtin_amdgcn_wmma_f32_16x16x32_bf16(
      /*neg_a=*/false, a, /*neg_b=*/false, b,
      /*c_mod=*/(short)0, c, /*reuse_a=*/false, /*reuse_b=*/false);
}

// ---------------------------------------------------------------------------
// Kernel 0: transpose + bf16-convert the three 256x256 weight matrices.
// W[c][cout] (row-major) -> Wt[cout][c] bf16 so projection fragments are
// contiguous per-lane 32B loads.
// ---------------------------------------------------------------------------
__global__ __launch_bounds__(256) void prep_weights_k(
    const float* __restrict__ Wq, const float* __restrict__ Wk,
    const float* __restrict__ Wv,
    __bf16* __restrict__ Wtq, __bf16* __restrict__ Wtk,
    __bf16* __restrict__ Wtv) {
  int idx  = blockIdx.x * 256 + threadIdx.x;   // 0..65535
  int cout = idx >> 8;
  int c    = idx & 255;
  int src  = c * CCH + cout;
  Wtq[idx] = (__bf16)Wq[src];
  Wtk[idx] = (__bf16)Wk[src];
  Wtv[idx] = (__bf16)Wv[src];
}

// ---------------------------------------------------------------------------
// Kernel 1: projections.  Per workgroup: one slice, 64 spatial rows.
//   Q[p][cout], K[p][cout]  (row-major bf16)   = X  @ W + b
//   Vt[cout][p]             (transposed bf16)  = (Xd @ Wv + bv)^T
// X is staged channel-major->row-major through LDS as bf16.
// Block = 128 threads (4 waves). Dynamic LDS = 2 * 64 * 264 bf16.
// ---------------------------------------------------------------------------
#define XPITCH 264

__global__ __launch_bounds__(128) void proj_k(
    const float* __restrict__ img, const float* __restrict__ depth,
    const __bf16* __restrict__ Wtq, const __bf16* __restrict__ Wtk,
    const __bf16* __restrict__ Wtv,
    const float* __restrict__ bq, const float* __restrict__ bk,
    const float* __restrict__ bv,
    __bf16* __restrict__ Qw, __bf16* __restrict__ Kw,
    __bf16* __restrict__ Vtw) {
  extern __shared__ unsigned char dynsmem[];
  __bf16* Xi = (__bf16*)dynsmem;          // [64][XPITCH] img  rows (bf16)
  __bf16* Xd = Xi + 64 * XPITCH;          // [64][XPITCH] depth rows

  const int slice = blockIdx.x;
  const int pbase = blockIdx.y * 64;
  const int tid   = threadIdx.x;
  const int wave  = tid >> 5;
  const int lane  = tid & 31;
  const int l15   = lane & 15;
  const int hi    = lane >> 4;            // 0: K 0..15, 1: K 16..31

  // ---- Stage inputs: global [c][p] f32 (coalesced in p) -> LDS [p][c] bf16
  {
    const float* ib = img   + (size_t)slice * CCH * HW + pbase;
    const float* db = depth + (size_t)slice * CCH * HW + pbase;
#pragma unroll 4
    for (int it = 0; it < 128; ++it) {    // 64*256 / 128 threads
      int idx = it * 128 + tid;
      int c   = idx >> 6;
      int p   = idx & 63;
      Xi[p * XPITCH + c] = (__bf16)ib[(size_t)c * HW + p];
      Xd[p * XPITCH + c] = (__bf16)db[(size_t)c * HW + p];
    }
  }
  __syncthreads();

  // ---- Q and K GEMMs: wave w owns rows [w*16, w*16+16)
  v16bf xiF[8], xdF[8];
#pragma unroll
  for (int kc = 0; kc < 8; ++kc) {
    const int off = (wave * 16 + l15) * XPITCH + kc * 32 + hi * 16;
    xiF[kc] = load_frag(Xi + off);
    xdF[kc] = load_frag(Xd + off);
  }

  for (int ct = 0; ct < 16; ++ct) {       // 16 cout tiles
    float biasq = bq[ct * 16 + l15];
    float biask = bk[ct * 16 + l15];
    v8f aq, ak;
#pragma unroll
    for (int i = 0; i < 8; ++i) { aq[i] = biasq; ak[i] = biask; }
#pragma unroll
    for (int kc = 0; kc < 8; ++kc) {
      const int woff = (ct * 16 + l15) * CCH + kc * 32 + hi * 16;
      aq = wmma_bf16(xiF[kc], load_frag(Wtq + woff), aq);
      ak = wmma_bf16(xdF[kc], load_frag(Wtk + woff), ak);
    }
#pragma unroll
    for (int r = 0; r < 8; ++r) {
      size_t o = ((size_t)slice * HW + pbase + wave * 16 + r + hi * 8) * CCH +
                 ct * 16 + l15;
      Qw[o] = (__bf16)aq[r];
      Kw[o] = (__bf16)ak[r];
    }
  }

  // ---- Vt GEMM: M = cout (wave owns 64 couts), N = p (all 64 rows)
  for (int mt = 0; mt < 4; ++mt) {
    const int cbase = wave * 64 + mt * 16;
    v16bf wvF[8];
#pragma unroll
    for (int kc = 0; kc < 8; ++kc)
      wvF[kc] = load_frag(Wtv + (cbase + l15) * CCH + kc * 32 + hi * 16);
    float bvr[8];
#pragma unroll
    for (int r = 0; r < 8; ++r) bvr[r] = bv[cbase + r + hi * 8];

    for (int nt = 0; nt < 4; ++nt) {
      v8f av;
#pragma unroll
      for (int r = 0; r < 8; ++r) av[r] = bvr[r];
#pragma unroll
      for (int kc = 0; kc < 8; ++kc) {
        v16bf xb = load_frag(Xd + (nt * 16 + l15) * XPITCH + kc * 32 + hi * 16);
        av = wmma_bf16(wvF[kc], xb, av);
      }
#pragma unroll
      for (int r = 0; r < 8; ++r) {
        size_t o = ((size_t)slice * CCH + cbase + r + hi * 8) * HW + pbase +
                   nt * 16 + l15;
        Vtw[o] = (__bf16)av[r];
      }
    }
  }
}

// ---------------------------------------------------------------------------
// Kernel 2: attention + residual + channel-major store.
// Workgroup = 2 waves = 32 query rows of one slice; each wave owns 16 rows.
// Scores strip [32][1028] f32 lives in LDS (exact softmax, two passes,
// no online rescaling). K/V fragments stream from L2-resident bf16 ws.
// ---------------------------------------------------------------------------
#define AROWS 32
#define SCP   1028     // scores pitch in floats (bank-stagger)

__global__ __launch_bounds__(64) void attn_k(
    const __bf16* __restrict__ Qw, const __bf16* __restrict__ Kw,
    const __bf16* __restrict__ Vtw, const float* __restrict__ img,
    float* __restrict__ out) {
  extern __shared__ unsigned char dynsmem[];
  float* smem   = (float*)dynsmem;
  float* scores = smem;                       // [AROWS][SCP]
  float* rowmax = smem + AROWS * SCP;         // [AROWS]
  float* rowsum = rowmax + AROWS;             // [AROWS]

  const int slice = blockIdx.x;
  const int pbase = blockIdx.y * AROWS;
  const int tid   = threadIdx.x;
  const int wave  = tid >> 5;
  const int lane  = tid & 31;
  const int l15   = lane & 15;
  const int hi    = lane >> 4;
  const int wrow  = wave * 16;

  const __bf16* Qs = Qw  + (size_t)slice * HW * CCH;
  const __bf16* Ks = Kw  + (size_t)slice * HW * CCH;
  const __bf16* Vs = Vtw + (size_t)slice * CCH * HW;

  // Register-resident Q fragments for this wave's 16 rows (8 K-chunks).
  v16bf qf[8];
  {
    const __bf16* qrow = Qs + (size_t)(pbase + wrow + l15) * CCH + hi * 16;
#pragma unroll
    for (int kc = 0; kc < 8; ++kc) qf[kc] = load_frag(qrow + kc * 32);
  }

  // ---- Phase 1: S = scale * Q K^T -> LDS; per-lane running row max.
  float rmax[8];
#pragma unroll
  for (int r = 0; r < 8; ++r) rmax[r] = -3.0e38f;

  for (int nt = 0; nt < 64; ++nt) {           // 1024 keys / 16
    v8f acc = {0.f, 0.f, 0.f, 0.f, 0.f, 0.f, 0.f, 0.f};
    const __bf16* krow = Ks + (size_t)(nt * 16 + l15) * CCH + hi * 16;
#pragma unroll
    for (int kc = 0; kc < 8; ++kc)
      acc = wmma_bf16(qf[kc], load_frag(krow + kc * 32), acc);
#pragma unroll
    for (int r = 0; r < 8; ++r) {
      float v = acc[r] * SCALE;
      rmax[r] = fmaxf(rmax[r], v);
      scores[(wrow + r + hi * 8) * SCP + nt * 16 + l15] = v;
    }
  }
  // Finish row max: butterfly inside each 16-lane group, publish via LDS.
#pragma unroll
  for (int r = 0; r < 8; ++r) {
    float m = rmax[r];
    m = fmaxf(m, __shfl_xor(m, 1, 32));
    m = fmaxf(m, __shfl_xor(m, 2, 32));
    m = fmaxf(m, __shfl_xor(m, 4, 32));
    m = fmaxf(m, __shfl_xor(m, 8, 32));
    rmax[r] = m;
  }
  if (l15 == 0) {
#pragma unroll
    for (int r = 0; r < 8; ++r) rowmax[wrow + r + hi * 8] = rmax[r];
  }

  // ---- Phase 2: P = exp(S - max); O += P V  (16 cout tiles accumulate).
  const int   myrow  = wrow + l15;            // this lane's row (A layout)
  const float myrmax = rowmax[myrow];         // same-wave LDS dependency
  float mysum = 0.f;

  v8f oacc[16];
#pragma unroll
  for (int ct = 0; ct < 16; ++ct)
#pragma unroll
    for (int i = 0; i < 8; ++i) oacc[ct][i] = 0.f;

  for (int nc = 0; nc < 32; ++nc) {           // key chunks of 32
    const float* srow = &scores[myrow * SCP + nc * 32 + hi * 16];
    float ev[16];
#pragma unroll
    for (int j = 0; j < 4; ++j) {
      float4 s = *(const float4*)(srow + 4 * j);
      ev[4 * j + 0] = __expf(s.x - myrmax);
      ev[4 * j + 1] = __expf(s.y - myrmax);
      ev[4 * j + 2] = __expf(s.z - myrmax);
      ev[4 * j + 3] = __expf(s.w - myrmax);
    }
    FragBF pf;
    __bf16* pb = (__bf16*)&pf.v;
#pragma unroll
    for (int j = 0; j < 16; ++j) { mysum += ev[j]; pb[j] = (__bf16)ev[j]; }

#pragma unroll
    for (int ct = 0; ct < 16; ++ct) {
      const __bf16* vrow =
          Vs + (size_t)(ct * 16 + l15) * HW + nc * 32 + hi * 16;
      oacc[ct] = wmma_bf16(pf.v, load_frag(vrow), oacc[ct]);
    }
  }
  mysum += __shfl_xor(mysum, 16, 32);         // join the two half-rows
  if (lane < 16) rowsum[myrow] = mysum;

  // ---- Phase 3: normalize, LDS transpose (reuse scores region), residual.
  __syncthreads();                            // everyone done reading scores
  float* ost = smem;                          // overlay: [AROWS][257]
  float inv[8];
#pragma unroll
  for (int r = 0; r < 8; ++r) inv[r] = 1.0f / rowsum[wrow + r + hi * 8];
#pragma unroll
  for (int ct = 0; ct < 16; ++ct)
#pragma unroll
    for (int r = 0; r < 8; ++r)
      ost[(wrow + r + hi * 8) * 257 + ct * 16 + l15] = oacc[ct][r] * inv[r];
  __syncthreads();

  const float* imgs = img + (size_t)slice * CCH * HW + pbase;
  float*       outs = out + (size_t)slice * CCH * HW + pbase;
#pragma unroll 4
  for (int it = 0; it < 128; ++it) {          // 32*256 / 64 threads
    int idx = it * 64 + tid;
    int c   = idx >> 5;
    int p   = idx & 31;
    outs[(size_t)c * HW + p] = ost[p * 257 + c] + imgs[(size_t)c * HW + p];
  }
}

// ---------------------------------------------------------------------------
// Launch: prep -> proj -> attn, all on `stream` (in-order).
// Workspace (bf16): Wt q/k/v (3*128KB), Q (16MB), K (16MB), Vt (16MB).
// ---------------------------------------------------------------------------
extern "C" void kernel_launch(void* const* d_in, const int* in_sizes, int n_in,
                              void* d_out, int out_size, void* d_ws,
                              size_t ws_size, hipStream_t stream) {
  (void)in_sizes; (void)n_in; (void)out_size; (void)ws_size;
  const float* img   = (const float*)d_in[0];
  const float* depth = (const float*)d_in[1];
  const float* Wq    = (const float*)d_in[2];
  const float* bq    = (const float*)d_in[3];
  const float* Wk    = (const float*)d_in[4];
  const float* bk    = (const float*)d_in[5];
  const float* Wv    = (const float*)d_in[6];
  const float* bv    = (const float*)d_in[7];
  float* out = (float*)d_out;

  char* ws = (char*)d_ws;
  const size_t WT_BYTES = (size_t)CCH * CCH * 2;       // 128 KB each
  const size_t M_BYTES  = (size_t)NSLICE * HW * CCH * 2;  // 16 MB each
  __bf16* Wtq = (__bf16*)(ws);
  __bf16* Wtk = (__bf16*)(ws + WT_BYTES);
  __bf16* Wtv = (__bf16*)(ws + 2 * WT_BYTES);
  __bf16* Qw  = (__bf16*)(ws + 3 * WT_BYTES);
  __bf16* Kw  = (__bf16*)(ws + 3 * WT_BYTES + M_BYTES);
  __bf16* Vtw = (__bf16*)(ws + 3 * WT_BYTES + 2 * M_BYTES);

  prep_weights_k<<<CCH * CCH / 256, 256, 0, stream>>>(Wq, Wk, Wv, Wtq, Wtk,
                                                      Wtv);

  const size_t proj_lds = (size_t)2 * 64 * XPITCH * sizeof(__bf16);
  proj_k<<<dim3(NSLICE, HW / 64), 128, proj_lds, stream>>>(
      img, depth, Wtq, Wtk, Wtv, bq, bk, bv, Qw, Kw, Vtw);

  const size_t attn_lds = (size_t)(AROWS * SCP + 2 * AROWS) * sizeof(float);
  attn_k<<<dim3(NSLICE, HW / AROWS), 64, attn_lds, stream>>>(Qw, Kw, Vtw, img,
                                                             out);
}
